// DGCNN_7000796692674
// MI455X (gfx1250) — compile-verified, matrix operand order checked by reading, not run on hardware
//
#include <hip/hip_runtime.h>
#include <hip/hip_bf16.h>

// ---------------- problem constants (from reference) ----------------
enum : int {
    NUM_GRAPHS = 1024,
    P          = 128,              // nodes per graph
    DEG        = 16,               // avg degree
    NNODES     = NUM_GRAPHS * P,   // 131072
    EPG        = P * DEG,          // 2048 edges per graph (contiguous: sorted by src)
    NEDGES     = NUM_GRAPHS * EPG, // 2097152
    TOPK       = 20,
    LATENT     = 97
};

typedef __attribute__((ext_vector_type(16))) _Float16 v16h;
typedef __attribute__((ext_vector_type(8)))  float    v8f;

union Frag16 {            // one 16x32 f16 WMMA operand, per-lane view
    v16h v;
    struct { uint4 lo, hi; } u;   // halves 0..7 / 8..15 : two 16B chunks
};

// =====================================================================
// Kernel 1: per-graph edge prep (LDS float atomics)
//   x_edge[n] = sum_{row==n} edge_attr ;  dis[n] = rsqrt(1 + #(col==n))
// =====================================================================
__global__ void edge_prep_kernel(const float* __restrict__ edge_attr,
                                 const int* __restrict__ row,
                                 const int* __restrict__ col,
                                 float* __restrict__ x_edge,
                                 float* __restrict__ dis) {
    int g = blockIdx.x;
    int base = g * P;
    __shared__ float sXe[P];
    __shared__ float sDeg[P];
    for (int i = threadIdx.x; i < P; i += blockDim.x) { sXe[i] = 0.f; sDeg[i] = 0.f; }
    __syncthreads();
    int e0 = g * EPG;
    for (int e = threadIdx.x; e < EPG; e += blockDim.x) {
        int r = row[e0 + e] - base;
        int c = col[e0 + e] - base;
        atomicAdd(&sXe[r], edge_attr[e0 + e]);
        atomicAdd(&sDeg[c], 1.0f);
    }
    __syncthreads();
    for (int i = threadIdx.x; i < P; i += blockDim.x) {
        x_edge[base + i] = sXe[i];
        dis[base + i]    = rsqrtf(1.0f + sDeg[i]);
    }
}

// =====================================================================
// Kernel 2: weight convert: Bt[Npad][Kpad] (f16) = B[k][n] zero-padded.
// Transposed so that a B-matrix WMMA fragment is two contiguous 16B loads.
// =====================================================================
__global__ void weight_to_f16t_kernel(const float* __restrict__ B, int ldb, int kB, int Nact,
                                      _Float16* __restrict__ Bt, int Kpad, int Npad) {
    int i = blockIdx.x * blockDim.x + threadIdx.x;
    if (i >= Npad * Kpad) return;
    int n = i / Kpad, k = i % Kpad;
    float v = (k < kB && n < Nact) ? B[(size_t)k * ldb + n] : 0.f;
    Bt[i] = (_Float16)v;
}

// =====================================================================
// Kernel 3: generic WMMA GEMM, all-f16 unguarded b128 fragment loads.
//   C[M,Nact] = act(A[M,Kpad] @ Bt^T + bias), optional f16 copy C16.
// A row-major f16 stride Kpad; Bt row-major f16 [Npad][Kpad].
// One wave per 16x16 tile, 4 waves/block; wave-uniform exit only.
// =====================================================================
__global__ void gemm_wmma_f16(const _Float16* __restrict__ A,
                              const _Float16* __restrict__ Bt,
                              const float* __restrict__ bias, int relu,
                              float* __restrict__ C, _Float16* __restrict__ C16, int ldc,
                              int M, int Npad, int Nact, int Kpad) {
    int wid  = threadIdx.x >> 5;
    int lane = threadIdx.x & 31;
    int tilesN = Npad >> 4;
    int tilesTotal = (M >> 4) * tilesN;
    int tile = blockIdx.x * 4 + wid;
    if (tile >= tilesTotal) return;              // wave-uniform
    int tm = tile / tilesN, tn = tile % tilesN;
    int m0 = tm << 4, n0 = tn << 4;
    int group = lane >> 4;                       // K half-group held by this lane
    int mr    = lane & 15;
    const _Float16* arow = A  + (size_t)(m0 + mr) * Kpad + 8 * group;
    const _Float16* brow = Bt + (size_t)(n0 + mr) * Kpad + 8 * group;

    v8f acc = {};
    for (int kb = 0; kb < Kpad; kb += 32) {
        __builtin_prefetch(arow + kb + 64, 0, 1);
        Frag16 a, b;
        a.u.lo = *reinterpret_cast<const uint4*>(arow + kb);
        a.u.hi = *reinterpret_cast<const uint4*>(arow + kb + 16);
        b.u.lo = *reinterpret_cast<const uint4*>(brow + kb);
        b.u.hi = *reinterpret_cast<const uint4*>(brow + kb + 16);
        acc = __builtin_amdgcn_wmma_f32_16x16x32_f16(
                  false, a.v, false, b.v, (short)0, acc, false, false);
    }

    int n = n0 + mr;
    if (n < Nact) {
#pragma unroll
        for (int r = 0; r < 8; ++r) {
            int mm = m0 + r + 8 * group;         // D layout: vgpr r -> M = r + 8*group
            float v = acc[r];
            if (bias != nullptr) v += bias[n];
            if (relu) v = fmaxf(v, 0.f);
            C[(size_t)mm * ldc + n] = v;
            if (C16 != nullptr) C16[(size_t)mm * ldc + n] = (_Float16)v;
        }
    }
}

// =====================================================================
// Kernel 3b: GCN layer-1 GEMM: A = [x | x_edge | 0-pad], K=129 -> 160.
// k-tiles 0..3 stream x (f32, float4 loads + cvt); tile 4 is the
// wave-uniform special case (only k==128 nonzero -> x_edge).
// =====================================================================
__global__ void gemm_wmma_layer1(const float* __restrict__ x,
                                 const float* __restrict__ x_edge,
                                 const _Float16* __restrict__ Bt, // [32][160]
                                 float* __restrict__ C, int M) {
    const int Kpad = 160;
    int wid  = threadIdx.x >> 5;
    int lane = threadIdx.x & 31;
    const int tilesN = 2;                        // Npad = 32
    int tilesTotal = (M >> 4) * tilesN;
    int tile = blockIdx.x * 4 + wid;
    if (tile >= tilesTotal) return;              // wave-uniform
    int tm = tile / tilesN, tn = tile % tilesN;
    int m0 = tm << 4, n0 = tn << 4;
    int group = lane >> 4;
    int mr    = lane & 15;
    int m = m0 + mr;
    const float*    arow = x  + (size_t)m * 128 + 8 * group;
    const _Float16* brow = Bt + (size_t)(n0 + mr) * Kpad + 8 * group;

    v8f acc = {};
    for (int kb = 0; kb < Kpad; kb += 32) {      // uniform trip + uniform branch
        Frag16 b;
        b.u.lo = *reinterpret_cast<const uint4*>(brow + kb);
        b.u.hi = *reinterpret_cast<const uint4*>(brow + kb + 16);
        v16h av;
        if (kb < 128) {
            const float4* p0 = reinterpret_cast<const float4*>(arow + kb);
            const float4* p1 = reinterpret_cast<const float4*>(arow + kb + 16);
            float4 f0 = p0[0], f1 = p0[1];       // halves 0..7
            float4 f2 = p1[0], f3 = p1[1];       // halves 8..15
            av[0] = (_Float16)f0.x; av[1] = (_Float16)f0.y;
            av[2] = (_Float16)f0.z; av[3] = (_Float16)f0.w;
            av[4] = (_Float16)f1.x; av[5] = (_Float16)f1.y;
            av[6] = (_Float16)f1.z; av[7] = (_Float16)f1.w;
            av[8]  = (_Float16)f2.x; av[9]  = (_Float16)f2.y;
            av[10] = (_Float16)f2.z; av[11] = (_Float16)f2.w;
            av[12] = (_Float16)f3.x; av[13] = (_Float16)f3.y;
            av[14] = (_Float16)f3.z; av[15] = (_Float16)f3.w;
        } else {
            float e = x_edge[m];                 // unconditional load, valid m
#pragma unroll
            for (int h = 0; h < 16; ++h) av[h] = (_Float16)0.f;
            av[0] = (group == 0) ? (_Float16)e : (_Float16)0.f;  // k==128 only
        }
        acc = __builtin_amdgcn_wmma_f32_16x16x32_f16(
                  false, av, false, b.v, (short)0, acc, false, false);
    }

    int n = n0 + mr;
#pragma unroll
    for (int r = 0; r < 8; ++r) {
        int mm = m0 + r + 8 * group;
        C[(size_t)mm * 32 + n] = acc[r];
    }
}

// =====================================================================
// Kernel 4: GCN aggregation (per graph, LDS atomics); also emits f16 copy
//   out[c] = sum_{e:(r->c)} t[r]*dis[r]*dis[c] + t[c]*dis[c]^2 + bias
// =====================================================================
__global__ void gcn_aggregate_kernel(const float* __restrict__ t,
                                     const float* __restrict__ dis,
                                     const int* __restrict__ row,
                                     const int* __restrict__ col,
                                     const float* __restrict__ bias,
                                     float* __restrict__ out,
                                     _Float16* __restrict__ out16, int F) {
    int g = blockIdx.x;
    int base = g * P;
    __shared__ float sT[P * 32];
    __shared__ float sAcc[P * 32];
    __shared__ float sDis[P];
    for (int i = threadIdx.x; i < P * F; i += blockDim.x) {
        sT[i]   = t[(size_t)base * F + i];
        sAcc[i] = 0.f;
    }
    for (int i = threadIdx.x; i < P; i += blockDim.x) sDis[i] = dis[base + i];
    __syncthreads();
    int e0 = g * EPG;
    for (int e = threadIdx.x; e < EPG; e += blockDim.x) {
        int r = row[e0 + e] - base;
        int c = col[e0 + e] - base;
        float w = sDis[r] * sDis[c];
        for (int f = 0; f < F; ++f)
            atomicAdd(&sAcc[c * F + f], sT[r * F + f] * w);
    }
    __syncthreads();
    for (int i = threadIdx.x; i < P * F; i += blockDim.x) {
        int nloc = i / F, f = i % F;
        float sn = sDis[nloc] * sDis[nloc];
        float v = sAcc[i] + sT[i] * sn + bias[f];
        out[(size_t)base * F + i] = v;
        if (out16 != nullptr) out16[(size_t)base * F + i] = (_Float16)v;
    }
}

// =====================================================================
// Kernel 5: per-graph top-K (strict > => lowest index wins ties)
// =====================================================================
__global__ void topk_kernel(const float* __restrict__ o4, int* __restrict__ topk) {
    int g = blockIdx.x;
    __shared__ float s[P];
    for (int i = threadIdx.x; i < P; i += blockDim.x) s[i] = o4[g * P + i];
    __syncthreads();
    if (threadIdx.x == 0) {
        for (int kk = 0; kk < TOPK; ++kk) {
            int bi = 0; float bv = s[0];
            for (int i = 1; i < P; ++i) if (s[i] > bv) { bv = s[i]; bi = i; }
            topk[g * TOPK + kk] = bi;
            s[bi] = -3.4e38f;
        }
    }
}

// =====================================================================
// Kernel 6: conv head per graph -> dense[192] (f32 + f16 copy)
// =====================================================================
__global__ void head_kernel(const float* __restrict__ o1, const float* __restrict__ o2,
                            const float* __restrict__ o3, const float* __restrict__ o4,
                            const int* __restrict__ topk,
                            const float* __restrict__ c1_w, const float* __restrict__ c1_b,
                            const float* __restrict__ c2_w, const float* __restrict__ c2_b,
                            float* __restrict__ dense, _Float16* __restrict__ dense16) {
    int g = blockIdx.x;
    __shared__ float feat[TOPK][LATENT];
    __shared__ float c1[16][TOPK];
    __shared__ float mp[16][TOPK / 2];
    for (int i = threadIdx.x; i < TOPK * LATENT; i += blockDim.x) {
        int kk = i / LATENT, d = i % LATENT;
        int node = g * P + topk[g * TOPK + kk];
        float v;
        if (d < 32)      v = o1[(size_t)node * 32 + d];
        else if (d < 64) v = o2[(size_t)node * 32 + (d - 32)];
        else if (d < 96) v = o3[(size_t)node * 32 + (d - 64)];
        else             v = o4[node];
        feat[kk][d] = v;
    }
    __syncthreads();
    for (int i = threadIdx.x; i < 16 * TOPK; i += blockDim.x) {
        int c = i / TOPK, kk = i % TOPK;
        float acc = c1_b[c];
        for (int d = 0; d < LATENT; ++d) acc += feat[kk][d] * c1_w[c * LATENT + d];
        c1[c][kk] = fmaxf(acc, 0.f);
    }
    __syncthreads();
    for (int i = threadIdx.x; i < 16 * (TOPK / 2); i += blockDim.x) {
        int c = i / (TOPK / 2), t = i % (TOPK / 2);
        mp[c][t] = fmaxf(c1[c][2 * t], c1[c][2 * t + 1]);
    }
    __syncthreads();
    for (int i = threadIdx.x; i < 32 * 6; i += blockDim.x) {
        int o = i / 6, t = i % 6;
        float acc = c2_b[o];
        for (int ic = 0; ic < 16; ++ic)
            for (int kk = 0; kk < 5; ++kk)
                acc += mp[ic][t + kk] * c2_w[(o * 16 + ic) * 5 + kk];
        float v = fmaxf(acc, 0.f);
        dense  [(size_t)g * 192 + o * 6 + t] = v;
        dense16[(size_t)g * 192 + o * 6 + t] = (_Float16)v;
    }
}

// =====================================================================
// Kernel 7: row-wise log_softmax over 16 logits
// =====================================================================
__global__ void logsoftmax_kernel(const float* __restrict__ logits, float* __restrict__ out) {
    int g = blockIdx.x * blockDim.x + threadIdx.x;
    if (g >= NUM_GRAPHS) return;
    float mx = -3.4e38f;
    for (int i = 0; i < 16; ++i) mx = fmaxf(mx, logits[g * 16 + i]);
    float s = 0.f;
    for (int i = 0; i < 16; ++i) s += expf(logits[g * 16 + i] - mx);
    float ls = logf(s);
    for (int i = 0; i < 16; ++i) out[g * 16 + i] = logits[g * 16 + i] - mx - ls;
}

// ---------------------------------------------------------------------
static inline int gemm_blocks(int M, int Npad) {
    int tiles = (M / 16) * (Npad / 16);
    return (tiles + 3) / 4;
}

extern "C" void kernel_launch(void* const* d_in, const int* in_sizes, int n_in,
                              void* d_out, int out_size, void* d_ws, size_t ws_size,
                              hipStream_t stream) {
    (void)in_sizes; (void)n_in; (void)out_size; (void)ws_size;
    const float* x         = (const float*)d_in[0];   // [N,128]
    const float* edge_attr = (const float*)d_in[1];   // [E,1]
    const int*   eidx      = (const int*)  d_in[2];   // [2,E]
    const float* W1 = (const float*)d_in[3];  const float* b1 = (const float*)d_in[4];
    const float* W2 = (const float*)d_in[5];  const float* b2 = (const float*)d_in[6];
    const float* W3 = (const float*)d_in[7];  const float* b3 = (const float*)d_in[8];
    const float* W4 = (const float*)d_in[9];  const float* b4 = (const float*)d_in[10];
    const float* c1_w = (const float*)d_in[11]; const float* c1_b = (const float*)d_in[12];
    const float* c2_w = (const float*)d_in[13]; const float* c2_b = (const float*)d_in[14];
    const float* out_w = (const float*)d_in[15]; const float* out_b = (const float*)d_in[16];
    const float* h1_w  = (const float*)d_in[17]; const float* h1_b  = (const float*)d_in[18];
    const float* h2_w  = (const float*)d_in[19]; const float* h2_b  = (const float*)d_in[20];

    const int* row = eidx;
    const int* col = eidx + NEDGES;

    // workspace carve-up (256B aligned)
    char* w = (char*)d_ws;
    auto carve = [&](size_t bytes) { void* p = (void*)w; w += (bytes + 255) & ~(size_t)255; return p; };
    float*     x_edge = (float*)carve((size_t)NNODES * 4);
    float*     dis    = (float*)carve((size_t)NNODES * 4);
    float*     t      = (float*)carve((size_t)NNODES * 32 * 4);
    float*     o1     = (float*)carve((size_t)NNODES * 32 * 4);
    float*     o2     = (float*)carve((size_t)NNODES * 32 * 4);
    float*     o3     = (float*)carve((size_t)NNODES * 32 * 4);
    float*     o4     = (float*)carve((size_t)NNODES * 4);
    _Float16*  o1h    = (_Float16*)carve((size_t)NNODES * 32 * 2);
    _Float16*  o2h    = (_Float16*)carve((size_t)NNODES * 32 * 2);
    _Float16*  o3h    = (_Float16*)carve((size_t)NNODES * 32 * 2);
    int*       tk     = (int*)  carve((size_t)NUM_GRAPHS * TOPK * 4);
    float*     dense  = (float*)carve((size_t)NUM_GRAPHS * 192 * 4);
    _Float16*  denseh = (_Float16*)carve((size_t)NUM_GRAPHS * 192 * 2);
    float*     d1     = (float*)carve((size_t)NUM_GRAPHS * 1024 * 4);
    _Float16*  d1h    = (_Float16*)carve((size_t)NUM_GRAPHS * 1024 * 2);
    float*     d2     = (float*)carve((size_t)NUM_GRAPHS * 128 * 4);
    _Float16*  d2h    = (_Float16*)carve((size_t)NUM_GRAPHS * 128 * 2);
    float*     logits = (float*)carve((size_t)NUM_GRAPHS * 16 * 4);
    _Float16*  W1t    = (_Float16*)carve((size_t)32 * 160 * 2);
    _Float16*  W2t    = (_Float16*)carve((size_t)32 * 32 * 2);
    _Float16*  W3t    = (_Float16*)carve((size_t)32 * 32 * 2);
    _Float16*  W4t    = (_Float16*)carve((size_t)16 * 32 * 2);
    _Float16*  OWt    = (_Float16*)carve((size_t)1024 * 192 * 2);
    _Float16*  H1t    = (_Float16*)carve((size_t)128 * 1024 * 2);
    _Float16*  H2t    = (_Float16*)carve((size_t)16 * 128 * 2);
    float*     outp   = (float*)d_out;

    // 0) weight conversion (f16, transposed [Npad][Kpad], zero-padded)
    auto conv = [&](const float* B, int ldb, int kB, int Nact, _Float16* Bt, int Kpad, int Npad) {
        int total = Npad * Kpad;
        weight_to_f16t_kernel<<<(total + 255) / 256, 256, 0, stream>>>(B, ldb, kB, Nact, Bt, Kpad, Npad);
    };
    conv(W1, 32, 129, 32, W1t, 160, 32);
    conv(W2, 32, 32, 32, W2t, 32, 32);
    conv(W3, 32, 32, 32, W3t, 32, 32);
    conv(W4, 1, 32, 1, W4t, 32, 16);
    conv(out_w, 1024, 192, 1024, OWt, 192, 1024);
    conv(h1_w, 128, 1024, 128, H1t, 1024, 128);
    conv(h2_w, 16, 128, 16, H2t, 128, 16);

    // 1) edge prep: x_edge, dis
    edge_prep_kernel<<<NUM_GRAPHS, 256, 0, stream>>>(edge_attr, row, col, x_edge, dis);

    // 2) GCN layer 1: t = [x | x_edge] @ W1
    gemm_wmma_layer1<<<gemm_blocks(NNODES, 32), 128, 0, stream>>>(x, x_edge, W1t, t, NNODES);
    gcn_aggregate_kernel<<<NUM_GRAPHS, 256, 0, stream>>>(t, dis, row, col, b1, o1, o1h, 32);

    // 3) GCN layer 2
    gemm_wmma_f16<<<gemm_blocks(NNODES, 32), 128, 0, stream>>>(
        o1h, W2t, nullptr, 0, t, nullptr, 32, NNODES, 32, 32, 32);
    gcn_aggregate_kernel<<<NUM_GRAPHS, 256, 0, stream>>>(t, dis, row, col, b2, o2, o2h, 32);

    // 4) GCN layer 3
    gemm_wmma_f16<<<gemm_blocks(NNODES, 32), 128, 0, stream>>>(
        o2h, W3t, nullptr, 0, t, nullptr, 32, NNODES, 32, 32, 32);
    gcn_aggregate_kernel<<<NUM_GRAPHS, 256, 0, stream>>>(t, dis, row, col, b3, o3, o3h, 32);

    // 5) GCN layer 4 (out dim 1 padded to 16)
    gemm_wmma_f16<<<gemm_blocks(NNODES, 16), 128, 0, stream>>>(
        o3h, W4t, nullptr, 0, t, nullptr, 1, NNODES, 16, 1, 32);
    gcn_aggregate_kernel<<<NUM_GRAPHS, 256, 0, stream>>>(t, dis, row, col, b4, o4, nullptr, 1);

    // 6) per-graph top-K + conv head -> dense[G,192]
    topk_kernel<<<NUM_GRAPHS, 128, 0, stream>>>(o4, tk);
    head_kernel<<<NUM_GRAPHS, 256, 0, stream>>>(o1, o2, o3, o4, tk,
                                                c1_w, c1_b, c2_w, c2_b, dense, denseh);

    // 7) dense head GEMMs
    gemm_wmma_f16<<<gemm_blocks(NUM_GRAPHS, 1024), 128, 0, stream>>>(
        denseh, OWt, out_b, 1, d1, d1h, 1024, NUM_GRAPHS, 1024, 1024, 192);
    gemm_wmma_f16<<<gemm_blocks(NUM_GRAPHS, 128), 128, 0, stream>>>(
        d1h, H1t, h1_b, 1, d2, d2h, 128, NUM_GRAPHS, 128, 128, 1024);
    gemm_wmma_f16<<<gemm_blocks(NUM_GRAPHS, 16), 128, 0, stream>>>(
        d2h, H2t, h2_b, 0, logits, nullptr, 16, NUM_GRAPHS, 16, 16, 128);

    // 8) log_softmax -> output [G,16]
    logsoftmax_kernel<<<(NUM_GRAPHS + 255) / 256, 256, 0, stream>>>(logits, outp);
}
